// PSN_77455440216423
// MI455X (gfx1250) — matrix operand order repaired and checked by reading, not compiled, not run on hardware
//
#include <hip/hip_runtime.h>
#include <hip/hip_bf16.h>

// ---------------------------------------------------------------------------
// Problem constants (B=64, E=256, C=256, K=8192)
// ---------------------------------------------------------------------------
#define NROWS   16384          // B*E
#define CDIM    256            // channels (GEMM reduction dim)
#define KCB     8192           // codebook entries (GEMM N dim)
#define NOISE_STD 0.5f
#define BETA      0.25f
#define INV_TOTAL (1.0f / 4194304.0f)   // 1 / (B*E*C)

#define ROWS_PER_WAVE   32     // two 16-row A blocks -> each B frag feeds 2 WMMAs
#define WAVES_PER_BLOCK 8
#define ROWS_PER_BLOCK  (ROWS_PER_WAVE * WAVES_PER_BLOCK)   // 256
#define KGROUPS         4      // codebook split across blockIdx.y for occupancy
#define COLS_PER_GROUP  (KCB / KGROUPS)                     // 2048
#define COLS_PER_TILE   64
#define LDS_ROW         264    // ushorts per staged codebook row (256 + 16B pad)
#define TILE_USHORTS    (COLS_PER_TILE * LDS_ROW)           // 16896
#define NTILES_G        (COLS_PER_GROUP / COLS_PER_TILE)    // 32

typedef __bf16 v16bf __attribute__((ext_vector_type(16)));
typedef float  v8f   __attribute__((ext_vector_type(8)));

union BFrag {
    v16bf v;
    uint4 q[2];
};

__device__ __forceinline__ ushort f2bf(float f) {
    // round-to-nearest-even f32 -> bf16
    unsigned u = __float_as_uint(f);
    unsigned r = u + 0x7FFFu + ((u >> 16) & 1u);
    return (ushort)(r >> 16);
}

// ---------------------------------------------------------------------------
// Kernel 1: latents f32 -> bf16 (row-major, contiguous)
// ---------------------------------------------------------------------------
__global__ __launch_bounds__(256)
void psn_cvt_latents(const float* __restrict__ in, ushort* __restrict__ out, int n) {
    int i = (blockIdx.x * 256 + threadIdx.x) * 4;
    if (i + 3 < n) {
        float4 f = *(const float4*)(in + i);
        ushort4 o;
        o.x = f2bf(f.x); o.y = f2bf(f.y); o.z = f2bf(f.z); o.w = f2bf(f.w);
        *(ushort4*)(out + i) = o;
    }
}

// ---------------------------------------------------------------------------
// Kernel 2: codebook f32 -> bf16, per-entry squared norms, zero loss slot.
// ---------------------------------------------------------------------------
__global__ __launch_bounds__(256)
void psn_cvt_codebook(const float* __restrict__ cb, ushort* __restrict__ cbbf,
                      float* __restrict__ wnorm, float* __restrict__ loss_slot) {
    const int k = blockIdx.x;
    const int c = threadIdx.x;
    float x = cb[k * CDIM + c];
    cbbf[k * CDIM + c] = f2bf(x);
    float s = x * x;
    #pragma unroll
    for (int m = 16; m >= 1; m >>= 1) s += __shfl_xor(s, m, 32);
    __shared__ float red[WAVES_PER_BLOCK];
    if ((c & 31) == 0) red[c >> 5] = s;
    __syncthreads();
    if (c == 0) {
        float t = 0.f;
        #pragma unroll
        for (int w = 0; w < WAVES_PER_BLOCK; ++w) t += red[w];
        wnorm[k] = t;
        if (k == 0) *loss_slot = 0.0f;     // deterministic re-init every launch
    }
}

// ---------------------------------------------------------------------------
// Async staging: 8 global_load_async_to_lds_b128 per thread move one
// 64-row x 512B codebook tile into padded LDS; tracked by ASYNCcnt.
// ---------------------------------------------------------------------------
__device__ __forceinline__ void stage_tile_async(const ushort* __restrict__ cbbf,
                                                 int col_base, ushort* lds_buf,
                                                 int tid) {
    #pragma unroll
    for (int i = 0; i < 8; ++i) {
        int slot = tid + i * 256;            // 0..2047
        int r    = slot >> 5;                // codebook row 0..63
        int ch   = slot & 31;                // 16B chunk 0..31
        unsigned long long g =
            (unsigned long long)(const uint4*)((cbbf + (col_base + r) * CDIM) + ch * 8);
        // flat shared pointer: low 32 bits == LDS byte address (aperture rule)
        unsigned l = (unsigned)(unsigned long long)(uint4*)(lds_buf + r * LDS_ROW + ch * 8);
        asm volatile("global_load_async_to_lds_b128 %0, %1, off"
                     :: "v"(l), "v"(g) : "memory");
    }
}

__device__ __forceinline__ void wait_async_all() {
    asm volatile("s_wait_asynccnt 0x0" ::: "memory");
}

// ---------------------------------------------------------------------------
// Kernel 3: fused bf16-WMMA distance GEMM + row argmin over a codebook slice.
//   score(n,k) = ||w_k||^2 - 2 * x_n . w_k      (||x||^2 dropped: row-const)
// Each wave owns 32 rows (two A blocks, resident in 128 VGPRs) so every B
// fragment pulled from LDS feeds two WMMAs (512 B/WMMA instead of 1 KB).
// blockIdx.y selects one of 4 codebook column groups; per-row partial
// (val,idx) results are merged by psn_merge. Async-DMA double buffering,
// one barrier per 64-column tile, 4 concurrent accumulator chains.
// ---------------------------------------------------------------------------
__global__ __launch_bounds__(256)
void psn_argmin_wmma(const ushort* __restrict__ latbf,
                     const ushort* __restrict__ cbbf,
                     const float*  __restrict__ wnorm,
                     float* __restrict__ pval,
                     int*   __restrict__ pidx) {
    __shared__ __align__(16) ushort ldsB[2 * TILE_USHORTS];   // 66 KB, 2 buffers

    const int tid  = threadIdx.x;
    const int lane = tid & 31;
    const int wave = tid >> 5;
    const int half = lane >> 4;     // 0: lanes 0-15, 1: lanes 16-31
    const int l15  = lane & 15;
    const int grp  = blockIdx.y;
    const int col0 = grp * COLS_PER_GROUP;
    const int row_base = blockIdx.x * ROWS_PER_BLOCK + wave * ROWS_PER_WAVE;

    // ---- Resident A fragments: 2 row blocks x 16 rows x 256 ch.
    // ISA 16-bit A layout: lanes 0-15 row=l15 K{0..7,16..23}; lanes 16-31
    // row=l15 K{8..15,24..31} per 32-chunk.
    BFrag afrag[2][8];
    #pragma unroll
    for (int rb = 0; rb < 2; ++rb) {
        const ushort* aptr = latbf + (row_base + rb * 16 + l15) * CDIM;
        #pragma unroll
        for (int kc = 0; kc < 8; ++kc) {
            const ushort* p = aptr + kc * 32 + half * 8;
            afrag[rb][kc].q[0] = *(const uint4*)(p);
            afrag[rb][kc].q[1] = *(const uint4*)(p + 16);
        }
    }

    float bestv[2][8];
    int   besti[2][8];
    #pragma unroll
    for (int rb = 0; rb < 2; ++rb)
        #pragma unroll
        for (int r = 0; r < 8; ++r) { bestv[rb][r] = 3.4e38f; besti[rb][r] = 0; }

    // ---- Prologue: DMA tile 0 into buffer 0.
    stage_tile_async(cbbf, col0, ldsB, tid);
    wait_async_all();
    __syncthreads();

    #pragma unroll 1
    for (int t = 0; t < NTILES_G; ++t) {
        const int ct = col0 + t * COLS_PER_TILE;

        // Kick DMA for the next tile into the other buffer; overlaps compute.
        if (t + 1 < NTILES_G)
            stage_tile_async(cbbf, ct + COLS_PER_TILE,
                             ldsB + ((t + 1) & 1) * TILE_USHORTS, tid);

        const ushort* buf = ldsB + (t & 1) * TILE_USHORTS;

        // ---- Two subtile-pairs; each pair = 4 accumulator chains
        //      ([2 subtiles] x [2 row blocks]), kc outermost.
        #pragma unroll
        for (int sp = 0; sp < 2; ++sp) {
            const int scol = sp * 32;                    // subtile pair base
            float wn0 = wnorm[ct + scol + l15];
            float wn1 = wnorm[ct + scol + 16 + l15];

            v8f acc[2][2];                               // [subtile][rowblock]
            #pragma unroll
            for (int s = 0; s < 2; ++s)
                #pragma unroll
                for (int rb = 0; rb < 2; ++rb)
                    acc[s][rb] = (v8f){0.f,0.f,0.f,0.f,0.f,0.f,0.f,0.f};

            #pragma unroll
            for (int kc = 0; kc < 8; ++kc) {
                #pragma unroll
                for (int s = 0; s < 2; ++s) {
                    // B layout: lane = column (scol+s*16+l15); lanes 0-15
                    // K{0..7,16..23}, lanes 16-31 K{8..15,24..31} of this 32-chunk.
                    BFrag bf;
                    const ushort* bp = buf + (scol + s * 16 + l15) * LDS_ROW
                                           + kc * 32 + half * 8;
                    bf.q[0] = *(const uint4*)(bp);
                    bf.q[1] = *(const uint4*)(bp + 16);
                    // One LDS fragment -> two WMMAs (row blocks 0 and 1).
                    acc[s][0] = __builtin_amdgcn_wmma_f32_16x16x32_bf16(
                                    false, afrag[0][kc].v, false, bf.v,
                                    (short)0, acc[s][0], false, false);
                    acc[s][1] = __builtin_amdgcn_wmma_f32_16x16x32_bf16(
                                    false, afrag[1][kc].v, false, bf.v,
                                    (short)0, acc[s][1], false, false);
                }
            }

            // ---- Running argmin (ascending column order; strict < keeps
            //      the lowest index on ties).
            #pragma unroll
            for (int s = 0; s < 2; ++s) {
                const int mycol = ct + scol + s * 16 + l15;
                const float wn = s ? wn1 : wn0;
                #pragma unroll
                for (int rb = 0; rb < 2; ++rb) {
                    #pragma unroll
                    for (int r = 0; r < 8; ++r) {
                        float score = __builtin_fmaf(-2.0f, acc[s][rb][r], wn);
                        if (score < bestv[rb][r]) {
                            bestv[rb][r] = score; besti[rb][r] = mycol;
                        }
                    }
                }
            }
        }

        // Next tile's DMA must be complete & visible before anyone reads it.
        wait_async_all();
        __syncthreads();
    }

    // ---- Cross-lane argmin within each 16-lane half, then write partials.
    // C/D layout: lane l15 = column, half 0 -> rows 0..7, half 1 -> rows 8..15.
    #pragma unroll
    for (int rb = 0; rb < 2; ++rb) {
        #pragma unroll
        for (int r = 0; r < 8; ++r) {
            float v = bestv[rb][r];
            int   ii = besti[rb][r];
            #pragma unroll
            for (int m = 8; m >= 1; m >>= 1) {
                float ov = __shfl_xor(v, m, 32);
                int   oi = __shfl_xor(ii, m, 32);
                if (ov < v || (ov == v && oi < ii)) { v = ov; ii = oi; }
            }
            if (l15 == 0) {
                int prow = row_base + rb * 16 + half * 8 + r;
                pval[grp * NROWS + prow] = v;
                pidx[grp * NROWS + prow] = ii;
            }
        }
    }
}

// ---------------------------------------------------------------------------
// Kernel 3b: merge the KGROUPS partial argmins per row -> inds_det.
// ---------------------------------------------------------------------------
__global__ __launch_bounds__(256)
void psn_merge(const float* __restrict__ pval, const int* __restrict__ pidx,
               int* __restrict__ inds_det) {
    int row = blockIdx.x * 256 + threadIdx.x;
    if (row >= NROWS) return;
    float bv = pval[row];
    int   bi = pidx[row];
    #pragma unroll
    for (int g = 1; g < KGROUPS; ++g) {
        float v = pval[g * NROWS + row];
        int   i = pidx[g * NROWS + row];
        if (v < bv || (v == bv && i < bi)) { bv = v; bi = i; }
    }
    inds_det[row] = bi;
}

// ---------------------------------------------------------------------------
// Kernel 4: per-row epilogue. One 256-thread block per row.
//   offs = round(noise*0.5); inds_noisy = clip(inds_det+offs)
//   st = x + (q_noisy - x);  loss += beta*(x-q_det)^2 + (q_noisy-x)^2
// ---------------------------------------------------------------------------
__global__ __launch_bounds__(256)
void psn_epilogue(const float* __restrict__ lat,
                  const float* __restrict__ cb,
                  const float* __restrict__ noise,
                  const int*   __restrict__ inds_det,
                  float* __restrict__ st_out,
                  float* __restrict__ loss_slot,
                  int*   __restrict__ inds_noisy_out) {
    const int row = blockIdx.x;
    const int c   = threadIdx.x;

    const int id  = inds_det[row];
    int in_ = id + (int)rintf(noise[row] * NOISE_STD);   // rintf == round-half-even
    in_ = in_ < 0 ? 0 : (in_ > KCB - 1 ? KCB - 1 : in_);

    float x  = lat[row * CDIM + c];
    float qd = cb[id  * CDIM + c];
    float qn = cb[in_ * CDIM + c];

    st_out[row * CDIM + c] = x + (qn - x);               // straight-through

    float dqd = x - qd;
    float dqn = qn - x;
    float l = BETA * dqd * dqd + dqn * dqn;
    #pragma unroll
    for (int m = 16; m >= 1; m >>= 1) l += __shfl_xor(l, m, 32);
    __shared__ float red[WAVES_PER_BLOCK];
    if ((c & 31) == 0) red[c >> 5] = l;
    __syncthreads();
    if (c == 0) {
        float t = 0.f;
        #pragma unroll
        for (int w = 0; w < WAVES_PER_BLOCK; ++w) t += red[w];
        atomicAdd(loss_slot, t * INV_TOTAL);
        inds_noisy_out[row] = in_;
    }
}

// ---------------------------------------------------------------------------
// Host launcher
// d_in : [0] latents f32 (64,256,256)  [1] codebook f32 (8192,256)
//        [2] noise f32 (16384,1)
// d_out: [0 .. 4194304) st f32 | [4194304] loss | [4194305 ..) inds_noisy i32
// ---------------------------------------------------------------------------
extern "C" void kernel_launch(void* const* d_in, const int* in_sizes, int n_in,
                              void* d_out, int out_size, void* d_ws, size_t ws_size,
                              hipStream_t stream) {
    (void)in_sizes; (void)n_in; (void)out_size; (void)ws_size;

    const float* latents  = (const float*)d_in[0];
    const float* codebook = (const float*)d_in[1];
    const float* noise    = (const float*)d_in[2];

    float* st_out    = (float*)d_out;
    float* loss_slot = st_out + (size_t)NROWS * CDIM;          // 4194304
    int*   inds_out  = (int*)(loss_slot + 1);                  // 4194305

    // Workspace layout: latbf16 | cbbf16 | wnorm | pval | pidx | inds_det
    char* ws = (char*)d_ws;
    size_t off = 0;
    ushort* latbf    = (ushort*)(ws + off); off += (size_t)NROWS * CDIM * 2;   // 8 MB
    ushort* cbbf     = (ushort*)(ws + off); off += (size_t)KCB * CDIM * 2;     // 4 MB
    float*  wnorm    = (float*) (ws + off); off += (size_t)KCB * 4;            // 32 KB
    float*  pval     = (float*) (ws + off); off += (size_t)KGROUPS * NROWS * 4;// 256 KB
    int*    pidx     = (int*)   (ws + off); off += (size_t)KGROUPS * NROWS * 4;// 256 KB
    int*    inds_det = (int*)   (ws + off);                                    // 64 KB

    // 1) latents -> bf16 (4 elems/thread)
    psn_cvt_latents<<<(NROWS * CDIM) / (256 * 4), 256, 0, stream>>>(
        latents, latbf, NROWS * CDIM);

    // 2) codebook -> bf16 + norms + zero loss
    psn_cvt_codebook<<<KCB, 256, 0, stream>>>(codebook, cbbf, wnorm, loss_slot);

    // 3) fused WMMA distance GEMM + argmin over 4 codebook slices
    dim3 grid(NROWS / ROWS_PER_BLOCK, KGROUPS);                // 64 x 4
    psn_argmin_wmma<<<grid, 256, 0, stream>>>(latbf, cbbf, wnorm, pval, pidx);

    // 3b) merge slice partials
    psn_merge<<<(NROWS + 255) / 256, 256, 0, stream>>>(pval, pidx, inds_det);

    // 4) gather / ST output / loss / noisy indices
    psn_epilogue<<<NROWS, 256, 0, stream>>>(
        latents, codebook, noise, inds_det, st_out, loss_slot, inds_out);
}
